// NodeReadout_12429635354784
// MI455X (gfx1250) — matrix operand order repaired and checked by reading, not compiled, not run on hardware
//
#include <hip/hip_runtime.h>

#define D 64
#define WT_STRIDE 136   // 128 bf16 + 8 pad: lane N -> banks 4N..4N+3, conflict-free

typedef __attribute__((ext_vector_type(16))) __bf16 v16bf;
typedef __attribute__((ext_vector_type(8)))  __bf16 v8bf;
typedef __attribute__((ext_vector_type(8)))  float  v8f;

// ---------------- kernel 1: zero the aggregation buffer (workspace) ----------
__global__ __launch_bounds__(256) void nr_zero_kernel(float4* p, int n4) {
    int i = blockIdx.x * blockDim.x + threadIdx.x;
    if (i < n4) p[i] = make_float4(0.f, 0.f, 0.f, 0.f);
}

// ---------------- kernel 2: scatter-add edges onto nodes ---------------------
// 16 threads per edge; each thread moves one float4 (coalesced 256B per edge)
// and issues 4 hardware global_atomic_add_f32 into L2-resident agg.
__global__ __launch_bounds__(256) void nr_scatter_kernel(const float* __restrict__ es,
                                                         const int*   __restrict__ dst,
                                                         float* __restrict__ agg,
                                                         int n_edges) {
    int gid = blockIdx.x * 256 + threadIdx.x;
    int e = gid >> 4;
    if (e >= n_edges) return;
    int c = (gid & 15) * 4;
    int d = dst[e];
    const float4 v = *(const float4*)(es + (size_t)e * D + c);
    float* a = agg + (size_t)d * D + c;
    unsafeAtomicAdd(a + 0, v.x);
    unsafeAtomicAdd(a + 1, v.y);
    unsafeAtomicAdd(a + 2, v.z);
    unsafeAtomicAdd(a + 3, v.w);
}

// ---------------- kernel 3: fused concat-GEMM + bias + ReLU via WMMA ---------
// out[16-tile, 0:64] = relu([nf | agg] @ W + b), K=128 split as 2 sources x 2
// k-steps of v_wmma_f32_16x16x32_bf16. One wave per 16-node tile, 4 N-tiles.
__global__ __launch_bounds__(256) void nr_gemm_kernel(const float* __restrict__ nf,
                                                      const float* __restrict__ agg,
                                                      const float* __restrict__ W,
                                                      const float* __restrict__ bias,
                                                      float* __restrict__ out,
                                                      int n_tiles) {
    __shared__ __bf16 ldsWt[D * WT_STRIDE];   // W^T in bf16: [n][k], padded rows

    const int tid = threadIdx.x;
    // Cooperatively stage W (128x64 f32, row-major) transposed into LDS as bf16.
    for (int idx = tid; idx < 2 * D * D; idx += 256) {
        int k = idx >> 6;        // 0..127
        int n = idx & (D - 1);   // 0..63
        ldsWt[n * WT_STRIDE + k] = (__bf16)W[idx];
    }
    __syncthreads();

    const int wave = tid >> 5;
    const int lane = tid & 31;
    const int tile = blockIdx.x * 8 + wave;
    if (tile >= n_tiles) return;

    const int h = lane >> 4;          // half-wave select
    const int N = lane & 15;          // column-within-tile / A-row
    const int node_base = tile * 16;
    const int arow = node_base + N;   // this lane's A-matrix row

    v8f acc[4] = {v8f{}, v8f{}, v8f{}, v8f{}};

    const float* srcs[2] = { nf, agg };
    #pragma unroll
    for (int s = 0; s < 2; ++s) {
        const float* rowp = srcs[s] + (size_t)arow * D;
        #pragma unroll
        for (int kk = 0; kk < D; kk += 32) {
            // Pack A (16x32 bf16): VGPR0-7 hold K=16h+0..15
            const int cb = kk + 16 * h;
            v16bf a;
            #pragma unroll
            for (int i = 0; i < 8; ++i) a[i]     = (__bf16)rowp[cb + i];
            #pragma unroll
            for (int i = 0; i < 8; ++i) a[8 + i] = (__bf16)rowp[cb + 8 + i];

            // B (32x16 bf16): lane supplies K = kbase..kbase+15 for column N
            const int kbase = s * D + kk + 16 * h;
            #pragma unroll
            for (int nt = 0; nt < 4; ++nt) {
                const __bf16* bp = &ldsWt[(nt * 16 + N) * WT_STRIDE + kbase];
                v8bf blo = *(const v8bf*)bp;        // 16B aligned ds_load
                v8bf bhi = *(const v8bf*)(bp + 8);
                v16bf bm;
                #pragma unroll
                for (int i = 0; i < 8; ++i) { bm[i] = blo[i]; bm[8 + i] = bhi[i]; }
                acc[nt] = __builtin_amdgcn_wmma_f32_16x16x32_bf16(
                    /*neg_a=*/false, a, /*neg_b=*/false, bm,
                    /*c_mod=*/(short)0, acc[nt],
                    /*reuse_a=*/false, /*reuse_b=*/false);
            }
        }
    }

    // Epilogue: bias + ReLU, D-layout: VGPR v -> row v + 8h, col = nt*16 + N.
    #pragma unroll
    for (int nt = 0; nt < 4; ++nt) {
        const int col = nt * 16 + N;
        const float bv = bias[col];
        #pragma unroll
        for (int v = 0; v < 8; ++v) {
            float r = acc[nt][v] + bv;
            r = fmaxf(r, 0.0f);
            out[(size_t)(node_base + v + 8 * h) * D + col] = r;
        }
    }
}

extern "C" void kernel_launch(void* const* d_in, const int* in_sizes, int n_in,
                              void* d_out, int out_size, void* d_ws, size_t ws_size,
                              hipStream_t stream) {
    const float* nf  = (const float*)d_in[0];
    const float* es  = (const float*)d_in[1];
    const int*   dst = (const int*)d_in[2];
    const float* W   = (const float*)d_in[3];
    const float* b   = (const float*)d_in[4];
    float* out = (float*)d_out;
    float* agg = (float*)d_ws;        // n_nodes * D floats of scratch

    const int n_nodes = in_sizes[0] / D;
    const int n_edges = in_sizes[2];

    // 1) zero agg (ws is poisoned; must be re-zeroed every call)
    const int n4 = (n_nodes * D) / 4;
    nr_zero_kernel<<<(n4 + 255) / 256, 256, 0, stream>>>((float4*)agg, n4);

    // 2) scatter-add edge states
    const long sthreads = (long)n_edges * 16;
    nr_scatter_kernel<<<(int)((sthreads + 255) / 256), 256, 0, stream>>>(es, dst, agg, n_edges);

    // 3) fused concat-GEMM + bias + ReLU
    const int n_tiles = (n_nodes + 15) / 16;
    nr_gemm_kernel<<<(n_tiles + 7) / 8, 256, 0, stream>>>(nf, agg, W, b, out, n_tiles);
}
